// Attention_42743514530305
// MI455X (gfx1250) — compile-verified, matrix operand order checked by reading, not run on hardware
//
#include <hip/hip_runtime.h>
#include <hip/hip_bf16.h>
#include <math.h>

typedef __attribute__((ext_vector_type(16))) _Float16 v16h;
typedef __attribute__((ext_vector_type(8)))  float    v8f;

#define NBATCH 2
#define NHEAD  8
#define SEQ    4096       // 64*64
#define HDIM   32         // 256 / 8
#define CPLANE 256        // NHEAD*HDIM

// ---------------------------------------------------------------------------
// Preprocess 1: Q,K fp32 -> f16 (same layout).
// Softmax temperature AND log2(e) folded into Q so the softmax runs in the
// exp2 domain (v_exp_f32 with no preceding v_mul per score).
// ---------------------------------------------------------------------------
__global__ __launch_bounds__(256)
void qk_cvt_kernel(const float* __restrict__ q, const float* __restrict__ k,
                   _Float16* __restrict__ q16, _Float16* __restrict__ k16,
                   float qscale, int n) {
    for (int i = blockIdx.x * blockDim.x + threadIdx.x; i < n;
         i += gridDim.x * blockDim.x) {
        q16[i] = (_Float16)(q[i] * qscale);
        k16[i] = (_Float16)(k[i]);
    }
}

// ---------------------------------------------------------------------------
// Preprocess 2: V [n, s, h, d] fp32  ->  Vt [n, h, d, s] f16 (LDS-tiled transpose)
// grid = (SEQ/32, NHEAD, NBATCH), block = 256
// ---------------------------------------------------------------------------
__global__ __launch_bounds__(256)
void v_transpose_kernel(const float* __restrict__ v, _Float16* __restrict__ vt) {
    __shared__ _Float16 tile[32][33];
    const int nb = blockIdx.z, h = blockIdx.y, s0 = blockIdx.x * 32;

    {   // coalesced read along d
        const int d  = threadIdx.x & 31;
        const int sr = threadIdx.x >> 5;           // 0..7
        #pragma unroll
        for (int i = 0; i < 4; ++i) {
            const int s = s0 + sr + i * 8;
            tile[sr + i * 8][d] =
                (_Float16)v[((size_t)(nb * SEQ + s) * NHEAD + h) * HDIM + d];
        }
    }
    __syncthreads();
    {   // coalesced write along s
        const int s  = threadIdx.x & 31;
        const int dr = threadIdx.x >> 5;           // 0..7
        #pragma unroll
        for (int i = 0; i < 4; ++i) {
            const int d = dr + i * 8;
            vt[((size_t)(nb * NHEAD + h) * HDIM + d) * SEQ + s0 + s] = tile[s][d];
        }
    }
}

// ---------------------------------------------------------------------------
// Flash attention. One wave owns a 16-query tile of one (n, h).
// S^T = K x Q^T  (A = K 16x32 f16, B = Q^T 32x16 f16, C = 16s x 16q f32)
// outT += V^T x P^T  (A = V^T 16x32, B = P^T 32x16, C = 16d x 16q f32)
// block = 256 threads = 8 waves; grid = NBATCH*NHEAD*(SEQ/16)/8 = 512 blocks
// ---------------------------------------------------------------------------
__global__ __launch_bounds__(256)
void attention_kernel(const _Float16* __restrict__ q16,
                      const _Float16* __restrict__ k16,
                      const _Float16* __restrict__ vt16,
                      float* __restrict__ out) {
    const int wave  = threadIdx.x >> 5;
    const int lane  = threadIdx.x & 31;
    const int tid   = blockIdx.x * 8 + wave;        // 0..4095
    const int nh    = tid >> 8;                     // (n,h) pair 0..15
    const int qtile = tid & 255;                    // 0..255
    const int nb    = nh >> 3;
    const int h     = nh & 7;
    const int q0    = qtile * 16;

    const int lmod  = lane & 15;
    const int lhalf = lane >> 4;                    // 0 or 1
    const bool lowhalf = (lhalf == 0);

    // ---- B operand: Q^T 32x16, resident for the whole loop -----------------
    // lane<16: column q=lane, d=0..15 ; lane>=16: column q=lane-16, d=16..31
    const v16h bq = *(const v16h*)(q16 +
        ((size_t)(nb * SEQ + q0 + lmod) * NHEAD + h) * HDIM + lhalf * 16);

    v8f acc0 = {};          // outT rows d=0..15  (C layout)
    v8f acc1 = {};          // outT rows d=16..31
    float mrun = -INFINITY; // running max (log2 domain) of this query column
    float lrun = 0.0f;      // running denominator

    const size_t kbase  = ((size_t)nb * SEQ * NHEAD + h) * HDIM; // + s*NHEAD*HDIM
    const size_t vtbase = (size_t)(nb * NHEAD + h) * HDIM * SEQ; // + d*SEQ + s

    for (int s0 = 0; s0 < SEQ; s0 += 32) {
        // ---- scores: two 16x16 S^T tiles over 32 keys ----------------------
        const v16h a0 = *(const v16h*)(k16 + kbase +
            (size_t)(s0 + lmod) * CPLANE + lhalf * 16);
        const v16h a1 = *(const v16h*)(k16 + kbase +
            (size_t)(s0 + 16 + lmod) * CPLANE + lhalf * 16);
        v8f zero = {};
        v8f st0 = __builtin_amdgcn_wmma_f32_16x16x32_f16(
            false, a0, false, bq, (short)0, zero, false, false);
        v8f st1 = __builtin_amdgcn_wmma_f32_16x16x32_f16(
            false, a1, false, bq, (short)0, zero, false, false);

        // gather this lane's 16 s-values of its query column:
        // lane<16 owns s=0..7 (st0) and gets s=8..15 from partner;
        // lane>=16 owns s=24..31 (st1) and gets s=16..23 from partner.
        float x[8], y[8];
        #pragma unroll
        for (int r = 0; r < 8; ++r) {
            x[r] = lowhalf ? st0[r] : st1[r];
            float send = lowhalf ? st1[r] : st0[r];
            y[r] = __shfl_xor(send, 16, 32);
        }

        // ---- online softmax (log2 domain) ----------------------------------
        float bm = fmaxf(x[0], y[0]);
        #pragma unroll
        for (int r = 1; r < 8; ++r) bm = fmaxf(bm, fmaxf(x[r], y[r]));
        bm = fmaxf(bm, __shfl_xor(bm, 16, 32));

        // wave-uniform: rescale only when some column's max actually moved
        if (__any(bm > mrun)) {
            const float mnew = fmaxf(mrun, bm);
            const float corr = __builtin_amdgcn_exp2f(mrun - mnew); // 0 first time
            mrun = mnew;
            lrun *= corr;
            #pragma unroll
            for (int r = 0; r < 8; ++r) { acc0[r] *= corr; acc1[r] *= corr; }
        }

        float px[8], py[8], lsum = 0.0f;
        #pragma unroll
        for (int r = 0; r < 8; ++r) {
            px[r] = __builtin_amdgcn_exp2f(x[r] - mrun);
            py[r] = __builtin_amdgcn_exp2f(y[r] - mrun);
            lsum += px[r] + py[r];
        }
        lsum += __shfl_xor(lsum, 16, 32);
        lrun += lsum;

        // ---- pack P^T into B-operand layout --------------------------------
        // lane<16 : k slots 0..15 = s 0..15  -> [px, py]
        // lane>=16: k slots 0..15 = s 16..31 -> [py, px]
        v16h bp;
        #pragma unroll
        for (int r = 0; r < 8; ++r) {
            bp[r]     = (_Float16)(lowhalf ? px[r] : py[r]);
            bp[r + 8] = (_Float16)(lowhalf ? py[r] : px[r]);
        }

        // ---- numerator: outT += V^T x P^T ----------------------------------
        const v16h av0 = *(const v16h*)(vt16 + vtbase +
            (size_t)lmod * SEQ + s0 + lhalf * 16);
        const v16h av1 = *(const v16h*)(vt16 + vtbase +
            (size_t)(16 + lmod) * SEQ + s0 + lhalf * 16);
        acc0 = __builtin_amdgcn_wmma_f32_16x16x32_f16(
            false, av0, false, bp, (short)0, acc0, false, false);
        acc1 = __builtin_amdgcn_wmma_f32_16x16x32_f16(
            false, av1, false, bp, (short)0, acc1, false, false);
    }

    // ---- epilogue: divide by denominator, write out [n, q, h, d] -----------
    const float inv = 1.0f / lrun;
    const int   qq  = q0 + lmod;
    float* obase = out + ((size_t)(nb * SEQ + qq) * NHEAD + h) * HDIM;
    const int dbase = lhalf * 8;                   // acc0 rows: d = dbase+r
    float4 w;
    w = make_float4(acc0[0]*inv, acc0[1]*inv, acc0[2]*inv, acc0[3]*inv);
    *(float4*)(obase + dbase)          = w;
    w = make_float4(acc0[4]*inv, acc0[5]*inv, acc0[6]*inv, acc0[7]*inv);
    *(float4*)(obase + dbase + 4)      = w;
    w = make_float4(acc1[0]*inv, acc1[1]*inv, acc1[2]*inv, acc1[3]*inv);
    *(float4*)(obase + 16 + dbase)     = w;
    w = make_float4(acc1[4]*inv, acc1[5]*inv, acc1[6]*inv, acc1[7]*inv);
    *(float4*)(obase + 16 + dbase + 4) = w;
}

// ---------------------------------------------------------------------------
extern "C" void kernel_launch(void* const* d_in, const int* in_sizes, int n_in,
                              void* d_out, int out_size, void* d_ws, size_t ws_size,
                              hipStream_t stream) {
    const float* q = (const float*)d_in[0];
    const float* k = (const float*)d_in[1];
    const float* v = (const float*)d_in[2];
    float* out = (float*)d_out;

    const int nelem = NBATCH * SEQ * CPLANE;       // 2,097,152 per tensor
    char* ws = (char*)d_ws;
    _Float16* q16  = (_Float16*)(ws);
    _Float16* k16  = (_Float16*)(ws + (size_t)nelem * 2);
    _Float16* vt16 = (_Float16*)(ws + (size_t)nelem * 4);

    // 1/sqrt(d) * log2(e): softmax computed with exp2
    const float qscale = (1.0f / sqrtf((float)HDIM)) * 1.44269504088896340736f;

    qk_cvt_kernel<<<2048, 256, 0, stream>>>(q, k, q16, k16, qscale, nelem);

    dim3 tgrid(SEQ / 32, NHEAD, NBATCH);
    v_transpose_kernel<<<tgrid, 256, 0, stream>>>(v, vt16);

    // 4096 wave-tiles, 8 waves/block
    attention_kernel<<<512, 256, 0, stream>>>(q16, k16, vt16, out);
}
